// _GATLayerV2_28484223107345
// MI455X (gfx1250) — matrix operand order repaired and checked by reading, not compiled, not run on hardware
//
#include <hip/hip_runtime.h>
#include <hip/hip_fp16.h>

typedef __attribute__((ext_vector_type(16))) _Float16 v16h;
typedef __attribute__((ext_vector_type(8)))  float    v8f;

#define B_   8
#define N_   1024
#define FIN_ 256
#define H_   8
#define F_   64
#define HF_  512
#define NEG_SLOPE 0.2f

union FragH {
  v16h v;
  unsigned short u[16];
  uint4 q[2];
};

__device__ __forceinline__ unsigned short f2h_bits(float f) {
  _Float16 h = (_Float16)f;
  return __builtin_bit_cast(unsigned short, h);
}
__device__ __forceinline__ float h_bits2f(unsigned short u) {
  return (float)__builtin_bit_cast(_Float16, u);
}

// ---------------------------------------------------------------------------
// Kernel 1: h = x @ W  ([8192,256] x [256,512]) -> fp16 hT in workspace,
// stored TRANSPOSED per head: hT[b][head][f][j]  (flat: (b*512 + col)*1024 + n,
// col = head*64 + f, n = node within batch). This makes the aggregation
// kernel's WMMA B-operand two contiguous b128 loads per lane.
// Block = 256 threads (8 waves), tile 64(M) x 32(N), K-chunks of 32.
// ---------------------------------------------------------------------------
__global__ __launch_bounds__(256)
void gemm_h_kernel(const float* __restrict__ x, const float* __restrict__ W,
                   unsigned short* __restrict__ h16T) {
  __shared__ __align__(16) unsigned short aT[64][40]; // [m][k], 80B rows (16B aligned)
  __shared__ __align__(16) unsigned short bT[32][40]; // [n][k], transposed W chunk

  const int m0 = blockIdx.x * 64;
  const int n0 = blockIdx.y * 32;
  const int t = threadIdx.x;
  const int lane = t & 31;
  const int wave = t >> 5;
  const int mw = wave & 3;   // M subtile 0..3
  const int nw = wave >> 2;  // N subtile 0..1
  const int half = lane >> 4;
  const int l16 = lane & 15;

  v8f acc = {0.f, 0.f, 0.f, 0.f, 0.f, 0.f, 0.f, 0.f};

  for (int k0 = 0; k0 < FIN_; k0 += 32) {
    // stage A chunk: x[m0..m0+63][k0..k0+31] as fp16, row-major [m][k]
    {
      const int cg = (t & 7) * 4;
      const int r = t >> 3; // 0..31
#pragma unroll
      for (int rr = 0; rr < 2; ++rr) {
        const int m = r + rr * 32;
        const float4 v = *(const float4*)&x[(size_t)(m0 + m) * FIN_ + k0 + cg];
        aT[m][cg + 0] = f2h_bits(v.x);
        aT[m][cg + 1] = f2h_bits(v.y);
        aT[m][cg + 2] = f2h_bits(v.z);
        aT[m][cg + 3] = f2h_bits(v.w);
      }
    }
    // stage B chunk: W[k0..k0+31][n0..n0+31] transposed to [n][k]
    {
      const int cg = (t & 7) * 4;
      const int k = t >> 3; // 0..31
      const float4 v = *(const float4*)&W[(size_t)(k0 + k) * HF_ + n0 + cg];
      bT[cg + 0][k] = f2h_bits(v.x);
      bT[cg + 1][k] = f2h_bits(v.y);
      bT[cg + 2][k] = f2h_bits(v.z);
      bT[cg + 3][k] = f2h_bits(v.w);
    }
    __syncthreads();

    // A fragment: row = mw*16 + l16; elems 0..7 -> K=klo+e, 8..15 -> K=16+klo+e-8
    FragH a, b;
    const int klo = half * 8;
    a.q[0] = *(const uint4*)&aT[mw * 16 + l16][klo];
    a.q[1] = *(const uint4*)&aT[mw * 16 + l16][16 + klo];
    // B fragment: col = nw*16 + l16; elems 0..7 -> K=klo+e, 8..15 -> K=16+klo+e-8
    b.q[0] = *(const uint4*)&bT[nw * 16 + l16][klo];
    b.q[1] = *(const uint4*)&bT[nw * 16 + l16][16 + klo];

    acc = __builtin_amdgcn_wmma_f32_16x16x32_f16(false, a.v, false, b.v,
                                                 (short)0, acc, false, false);
    __syncthreads();
  }

  // D layout: VGPR r -> row r + half*8, col l16. Store transposed.
  const int col = n0 + nw * 16 + l16; // head*64 + f
#pragma unroll
  for (int r = 0; r < 8; ++r) {
    const int m = m0 + mw * 16 + r + half * 8; // global node row b*1024 + n
    const int bb = m >> 10;
    const int n = m & (N_ - 1);
    h16T[(((size_t)(bb * HF_ + col)) << 10) + n] = f2h_bits(acc[r]);
  }
}

// ---------------------------------------------------------------------------
// Kernel 2: el[node,head] = dot(h[node,head,:], a_left[head,:]); same for er.
// One thread per node; loads from hT are coalesced across adjacent nodes.
// ---------------------------------------------------------------------------
__global__ __launch_bounds__(256)
void edge_logits_kernel(const unsigned short* __restrict__ h16T,
                        const float* __restrict__ a_left,
                        const float* __restrict__ a_right,
                        float* __restrict__ el, float* __restrict__ er) {
  const int node = blockIdx.x * blockDim.x + threadIdx.x; // 0..8191
  const int b = node >> 10;
  const int n = node & (N_ - 1);
  const unsigned short* hb = &h16T[((size_t)b * HF_) << 10];

  for (int head = 0; head < H_; ++head) {
    float sl = 0.f, sr = 0.f;
#pragma unroll 8
    for (int f = 0; f < F_; ++f) {
      const int col = head * F_ + f;
      const float hv = h_bits2f(hb[(((size_t)col) << 10) + n]);
      sl += hv * a_left[col];
      sr += hv * a_right[col];
    }
    el[node * H_ + head] = sl;
    er[node * H_ + head] = sr;
  }
}

// ---------------------------------------------------------------------------
// Kernel 3: online softmax stats per (b,i,head): m = max_j e_ij, l = sum exp(e-m)
// over valid (masked-in) j. One thread per (node, head); heads adjacent so the
// mask/prior row loads broadcast across 8 lanes and er loads coalesce.
// Modest unroll: latency-bound streaming loop, keep the code compact.
// ---------------------------------------------------------------------------
__global__ __launch_bounds__(256)
void softmax_stats_kernel(const int* __restrict__ adj,
                          const float* __restrict__ prior,
                          const float* __restrict__ el,
                          const float* __restrict__ er,
                          const float* __restrict__ edge_scale_p,
                          float* __restrict__ mrow, float* __restrict__ lrow) {
  const int g = blockIdx.x * blockDim.x + threadIdx.x; // B*N*H threads
  const int head = g & 7;
  const int node = g >> 3;     // b*N + i
  const int b = node >> 10;
  const float es = edge_scale_p[0];
  const float eli = el[node * H_ + head];
  const int* maskrow = &adj[(size_t)node * N_];
  const float* prow = &prior[(size_t)node * N_];
  const float* erb = &er[(size_t)b * N_ * H_ + head];

  float m = -__builtin_inff();
  float l = 0.f;
#pragma unroll 4
  for (int j = 0; j < N_; ++j) {
    if ((j & 63) == 0 && j + 64 < N_) {
      __builtin_prefetch(&maskrow[j + 64], 0, 0); // global_prefetch_b8
      __builtin_prefetch(&prow[j + 64], 0, 0);
    }
    if (maskrow[j] != 0) {
      float v = eli + erb[(size_t)j * H_];
      v = (v > 0.f) ? v : NEG_SLOPE * v;
      const float e = v + es * prow[j];
      if (e > m) {
        l = l * __expf(m - e) + 1.f;
        m = e;
      } else {
        l += __expf(e - m);
      }
    }
  }
  mrow[node * H_ + head] = m;
  lrow[node * H_ + head] = l;
}

// ---------------------------------------------------------------------------
// Kernel 4: h_new[b,i,head,:] = sum_j alpha_ij * h[b,j,head,:]  via WMMA.
// Block = 128 threads (4 waves), one (b, head, 64-row i-block). Each wave owns
// a 16-row i-tile and 4 accumulators (f-tiles of 16). alpha A-fragments are
// generated on the fly (flash-style, exp(e - m)); B-fragments come straight
// from the transposed fp16 hT with two contiguous b128 loads per lane
// (L2-resident). No LDS, no barriers. Epilogue scales rows by 1/l.
// j-loop unroll capped at 2: enough to overlap alpha VALU/exp generation with
// WMMA + loads of the adjacent iteration without blowing up the I$.
// ---------------------------------------------------------------------------
__global__ __launch_bounds__(128)
void attn_aggregate_kernel(const unsigned short* __restrict__ h16T,
                           const int* __restrict__ adj,
                           const float* __restrict__ prior,
                           const float* __restrict__ el,
                           const float* __restrict__ er,
                           const float* __restrict__ mrow,
                           const float* __restrict__ lrow,
                           const float* __restrict__ edge_scale_p,
                           float* __restrict__ out) {
  const int b = blockIdx.z;
  const int head = blockIdx.y;
  const int i0 = blockIdx.x * 64;
  const int t = threadIdx.x;
  const int lane = t & 31;
  const int wave = t >> 5;
  const int half = lane >> 4;
  const int l16 = lane & 15;

  const int i = i0 + wave * 16 + l16; // this lane's A-matrix row
  const int node = b * N_ + i;
  const float es = edge_scale_p[0];
  const float eli = el[node * H_ + head];
  const float mi = mrow[node * H_ + head];
  const int* maskrow = &adj[(size_t)node * N_];
  const float* prow = &prior[(size_t)node * N_];
  const float* erb = &er[(size_t)b * N_ * H_ + head];
  // hT base for this (b, head): rows are f (64), columns are j (1024)
  const unsigned short* hbase = &h16T[((size_t)(b * HF_ + head * F_)) << 10];

  v8f acc[4];
#pragma unroll
  for (int ft = 0; ft < 4; ++ft)
    acc[ft] = (v8f){0.f, 0.f, 0.f, 0.f, 0.f, 0.f, 0.f, 0.f};

  const int klo = half * 8;

#pragma unroll 2
  for (int j0 = 0; j0 < N_; j0 += 32) {
    if (j0 + 32 < N_) {
      __builtin_prefetch(&maskrow[j0 + 32], 0, 0);
      __builtin_prefetch(&prow[j0 + 32], 0, 0);
    }

    // alpha A-fragment: elems 0..7 -> j = j0+klo+e ; 8..15 -> j = j0+16+klo+e-8
    FragH a;
#pragma unroll
    for (int run = 0; run < 2; ++run) {
      const int jr = j0 + run * 16 + klo; // multiple of 8 -> 32B aligned
      const int4 mk0 = *(const int4*)&maskrow[jr];
      const int4 mk1 = *(const int4*)&maskrow[jr + 4];
      const float4 p0 = *(const float4*)&prow[jr];
      const float4 p1 = *(const float4*)&prow[jr + 4];
      const int mk[8] = {mk0.x, mk0.y, mk0.z, mk0.w, mk1.x, mk1.y, mk1.z, mk1.w};
      const float pp[8] = {p0.x, p0.y, p0.z, p0.w, p1.x, p1.y, p1.z, p1.w};
#pragma unroll
      for (int e = 0; e < 8; ++e) {
        float aval = 0.f;
        if (mk[e] != 0) {
          float v = eli + erb[(size_t)(jr + e) * H_];
          v = (v > 0.f) ? v : NEG_SLOPE * v;
          aval = __expf(v + es * pp[e] - mi);
        }
        a.u[run * 8 + e] = f2h_bits(aval);
      }
    }

    // 4 f-tiles: B fragment = 8 fp16 at hT[f][j0 + half*8 ...] and 8 at hT[f][j0 + 16 + half*8 ...]
#pragma unroll
    for (int ft = 0; ft < 4; ++ft) {
      const unsigned short* hf =
          hbase + (((size_t)(ft * 16 + l16)) << 10) + j0 + half * 8;
      FragH bf;
      bf.q[0] = *(const uint4*)&hf[0];
      bf.q[1] = *(const uint4*)&hf[16];
      acc[ft] = __builtin_amdgcn_wmma_f32_16x16x32_f16(false, a.v, false, bf.v,
                                                       (short)0, acc[ft], false,
                                                       false);
    }
  }

  // epilogue: scale each output row by 1/l and store f32 (row-major output)
#pragma unroll
  for (int r = 0; r < 8; ++r) {
    const int irow = i0 + wave * 16 + r + half * 8;
    const float linv = 1.f / lrow[(size_t)(b * N_ + irow) * H_ + head];
#pragma unroll
    for (int ft = 0; ft < 4; ++ft) {
      out[(size_t)(b * N_ + irow) * HF_ + head * F_ + ft * 16 + l16] =
          acc[ft][r] * linv;
    }
  }
}

// ---------------------------------------------------------------------------
// Launcher. Workspace layout (bytes):
//   [0, 8MB)            h16T : fp16 h, transposed [B][H][F][N]
//   [8MB, +256KB)       el   : [B*N*H] f32
//   [+256KB, +256KB)    er
//   [+256KB, +256KB)    m
//   [+256KB, +256KB)    l      (total 9 MB)
// ---------------------------------------------------------------------------
extern "C" void kernel_launch(void* const* d_in, const int* in_sizes, int n_in,
                              void* d_out, int out_size, void* d_ws, size_t ws_size,
                              hipStream_t stream) {
  const float* x          = (const float*)d_in[0];
  const int*   adj        = (const int*)d_in[1];
  const float* prior      = (const float*)d_in[2];
  const float* W          = (const float*)d_in[3];
  const float* a_left     = (const float*)d_in[4];
  const float* a_right    = (const float*)d_in[5];
  const float* edge_scale = (const float*)d_in[6];
  float* out = (float*)d_out;

  char* ws = (char*)d_ws;
  unsigned short* h16T = (unsigned short*)ws;
  float* el   = (float*)(ws + (size_t)8388608);
  float* er   = (float*)(ws + (size_t)8650752);
  float* mrow = (float*)(ws + (size_t)8912896);
  float* lrow = (float*)(ws + (size_t)9175040);

  gemm_h_kernel<<<dim3(128, 16, 1), 256, 0, stream>>>(x, W, h16T);
  edge_logits_kernel<<<32, 256, 0, stream>>>(h16T, a_left, a_right, el, er);
  softmax_stats_kernel<<<256, 256, 0, stream>>>(adj, prior, el, er, edge_scale,
                                                mrow, lrow);
  attn_aggregate_kernel<<<dim3(16, 8, 8), 128, 0, stream>>>(
      h16T, adj, prior, el, er, mrow, lrow, edge_scale, out);
}
